// KmeansAudioQuantizer_11922829214092
// MI455X (gfx1250) — matrix operand order repaired and checked by reading, not compiled, not run on hardware
//
#include <hip/hip_runtime.h>
#include <hip/hip_bf16.h>
#include <cfloat>

// Problem constants (reference: B=8, L=1024, C=1024, N=5000)
#define B_    8
#define L_    1024
#define C_    1024
#define N_    5000
#define ROWS  (B_ * L_)            // 8192 rows of z
#define NPAD  5056                 // N padded to multiple of 64
#define NSPLIT 8                   // split the N range across 8 block groups
#define CHUNKS64 (NPAD / 64)       // 79 chunks of 64 codes
#define KPANEL 128                 // K values staged per LDS panel
#define LDS_ROW 264                // ushorts per LDS row: 2*128 + 8 pad (528B -> bank-conflict free)

typedef __attribute__((ext_vector_type(16))) __bf16 v16bf;
typedef __attribute__((ext_vector_type(8)))  __bf16 v8bf;
typedef __attribute__((ext_vector_type(8)))  float  v8f;
typedef __attribute__((ext_vector_type(4)))  int    v4i;

// ---------- async-to-LDS (CDNA5) ----------
#if __has_builtin(__builtin_amdgcn_global_load_async_to_lds_b128)
typedef __attribute__((address_space(1))) v4i gas_v4i;   // global <4 x i32>
typedef __attribute__((address_space(3))) v4i las_v4i;   // LDS <4 x i32>
static __device__ __forceinline__ void async_lds_b128(const unsigned short* g, unsigned short* l) {
    __builtin_amdgcn_global_load_async_to_lds_b128(
        (gas_v4i*)(unsigned short*)g, (las_v4i*)l, 0, 0);
}
#else
static __device__ __forceinline__ void async_lds_b128(const unsigned short* g, unsigned short* l) {
    unsigned loff = (unsigned)(unsigned long long)(uintptr_t)l;  // LDS aperture: offset in addr[31:0]
    asm volatile("global_load_async_to_lds_b128 %0, %1, off"
                 :: "v"(loff), "v"((unsigned long long)(uintptr_t)g) : "memory");
}
#endif

#if __has_builtin(__builtin_amdgcn_s_wait_asynccnt)
#define WAIT_ASYNC(n) __builtin_amdgcn_s_wait_asynccnt(n)
#else
#define WAIT_ASYNC(n) asm volatile("s_wait_asynccnt %0" :: "i"(n) : "memory")
#endif

// ---------- helpers ----------
static __device__ __forceinline__ unsigned short f32_to_bf16_rne(float f) {
    unsigned u = __float_as_uint(f);
    unsigned r = 0x7FFFu + ((u >> 16) & 1u);
    return (unsigned short)((u + r) >> 16);
}

static __device__ __forceinline__ void split_bf16(float f, unsigned short& h, unsigned short& l) {
    unsigned short hb = f32_to_bf16_rne(f);
    float hf = __uint_as_float((unsigned)hb << 16);
    h = hb;
    l = f32_to_bf16_rne(f - hf);
}

// order-preserving float -> uint mapping packed with index (ties -> smallest idx)
static __device__ __forceinline__ unsigned long long pack_key(float v, int idx) {
    unsigned fb = __float_as_uint(v);
    unsigned ob = (fb & 0x80000000u) ? ~fb : (fb | 0x80000000u);
    return ((unsigned long long)ob << 32) | (unsigned)idx;
}

// A fragment (16x32 bf16): lane 0-15 row M=lane; elems 0..7 = K[half*8 ..], 8..15 = K[16+half*8 ..]
static __device__ __forceinline__ v16bf load_a_frag(const unsigned short* p) {
    v8bf x0 = *reinterpret_cast<const v8bf*>(p);
    v8bf x1 = *reinterpret_cast<const v8bf*>(p + 16);
    v16bf r;
#pragma unroll
    for (int i = 0; i < 8; ++i) { r[i] = x0[i]; r[i + 8] = x1[i]; }
    return r;
}

// B fragment (32x16 bf16): elems 0..7 = K[half*8 ..], 8..15 = K[16+half*8 ..]
static __device__ __forceinline__ v16bf load_b_frag(const unsigned short* p) {
    v8bf x0 = *reinterpret_cast<const v8bf*>(p);
    v8bf x1 = *reinterpret_cast<const v8bf*>(p + 16);
    v16bf r;
#pragma unroll
    for (int i = 0; i < 8; ++i) { r[i] = x0[i]; r[i + 8] = x1[i]; }
    return r;
}

static __device__ __forceinline__ v8f wmma_bf16(v16bf a, v16bf b, v8f c) {
    return __builtin_amdgcn_wmma_f32_16x16x32_bf16(false, a, false, b, (short)0, c, false, false);
}

// ---------- kernel 1: init argmin keys ----------
__global__ void vq_init_keys(unsigned long long* __restrict__ keys) {
    int i = blockIdx.x * blockDim.x + threadIdx.x;
    if (i < ROWS) keys[i] = 0xFFFFFFFFFFFFFFFFull;
}

// ---------- kernel 2: split z into bf16 hi/lo ----------
__global__ void vq_convert_z(const float* __restrict__ z,
                             unsigned short* __restrict__ zh,
                             unsigned short* __restrict__ zl) {
    size_t i = ((size_t)blockIdx.x * blockDim.x + threadIdx.x) * 4;
    float4 f = *reinterpret_cast<const float4*>(z + i);
    ushort4 h, l;
    split_bf16(f.x, h.x, l.x);
    split_bf16(f.y, h.y, l.y);
    split_bf16(f.z, h.z, l.z);
    split_bf16(f.w, h.w, l.w);
    *reinterpret_cast<ushort4*>(zh + i) = h;
    *reinterpret_cast<ushort4*>(zl + i) = l;
}

// ---------- kernel 3: split codebook + row norms (pad rows: zeros, c2=FLT_MAX) ----------
__global__ void vq_convert_cb(const float* __restrict__ cb,
                              unsigned short* __restrict__ ch,
                              unsigned short* __restrict__ cl,
                              float* __restrict__ c2) {
    __shared__ float red[256];
    const int n = blockIdx.x;
    const int tid = threadIdx.x;
    const size_t base = (size_t)n * C_ + (size_t)tid * 4;
    float ss = 0.0f;
    ushort4 h = {0, 0, 0, 0}, l = {0, 0, 0, 0};
    if (n < N_) {
        float4 f = *reinterpret_cast<const float4*>(cb + base);
        split_bf16(f.x, h.x, l.x);
        split_bf16(f.y, h.y, l.y);
        split_bf16(f.z, h.z, l.z);
        split_bf16(f.w, h.w, l.w);
        ss = f.x * f.x + f.y * f.y + f.z * f.z + f.w * f.w;
    }
    *reinterpret_cast<ushort4*>(ch + base) = h;
    *reinterpret_cast<ushort4*>(cl + base) = l;
    red[tid] = ss;
    __syncthreads();
#pragma unroll
    for (int s = 128; s > 0; s >>= 1) {
        if (tid < s) red[tid] += red[tid + s];
        __syncthreads();
    }
    if (tid == 0) c2[n] = (n < N_) ? red[0] : FLT_MAX;
}

// ---------- kernel 4: fused WMMA GEMM + argmin (LDS-staged B, async double buffer) ----------
// grid = (ROWS/32/8)*NSPLIT blocks of 256 threads (8 waves).
// Each wave: 2 M-tiles (32 rows) x 4 N-tiles (64 codes/chunk), 24 WMMAs per K-step.
// All 8 waves share the 64-code B panel staged in LDS via GLOBAL_LOAD_ASYNC_TO_LDS_B128.
__global__ __launch_bounds__(256) void vq_gemm_argmin(
    const unsigned short* __restrict__ zh, const unsigned short* __restrict__ zl,
    const unsigned short* __restrict__ ch, const unsigned short* __restrict__ cl,
    const float* __restrict__ c2, unsigned long long* __restrict__ keys) {
    __shared__ unsigned short sB[2][64 * LDS_ROW];  // [buf][code 0..63][plane 0..1][K 0..127] (+pad)

    const int tid  = threadIdx.x;
    const int wave = tid >> 5;
    const int lane = tid & 31;
    const int half = lane >> 4;
    const int l16  = lane & 15;
    const int job  = (blockIdx.x / NSPLIT) * 8 + wave;  // 0..255
    const int seg  = blockIdx.x % NSPLIT;
    const int mtile0 = job * 2;
    const int mtile1 = job * 2 + 1;

    const unsigned short* zr0h = zh + (size_t)(mtile0 * 16 + l16) * C_ + half * 8;
    const unsigned short* zr0l = zl + (size_t)(mtile0 * 16 + l16) * C_ + half * 8;
    const unsigned short* zr1h = zh + (size_t)(mtile1 * 16 + l16) * C_ + half * 8;
    const unsigned short* zr1l = zl + (size_t)(mtile1 * 16 + l16) * C_ + half * 8;

    float minv0[8], minv1[8];
    int   mini0[8], mini1[8];
#pragma unroll
    for (int r = 0; r < 8; ++r) {
        minv0[r] = FLT_MAX; mini0[r] = 0x7FFFFFFF;
        minv1[r] = FLT_MAX; mini1[r] = 0x7FFFFFFF;
    }

#pragma unroll 1
    for (int chk = seg; chk < CHUNKS64; chk += NSPLIT) {
        const int nbase = chk * 64;

        v8f acc[2][4];
#pragma unroll
        for (int mt = 0; mt < 2; ++mt)
#pragma unroll
            for (int nt = 0; nt < 4; ++nt)
#pragma unroll
                for (int i = 0; i < 8; ++i) acc[mt][nt][i] = 0.0f;

        // stage K-panel 0
        {
#pragma unroll
            for (int j = 0; j < 8; ++j) {
                int t = tid + j * 256, r = t >> 5, rem = t & 31, p = rem >> 4, s = rem & 15;
                const unsigned short* g = (p ? cl : ch) + (size_t)(nbase + r) * C_ + s * 8;
                async_lds_b128(g, &sB[0][r * LDS_ROW + p * 128 + s * 8]);
            }
        }

#pragma unroll 1
        for (int kp = 0; kp < C_ / KPANEL; ++kp) {   // 8 panels
            if (kp < C_ / KPANEL - 1) {              // prefetch next panel into other buffer
                const int kn = (kp + 1) * KPANEL;
#pragma unroll
                for (int j = 0; j < 8; ++j) {
                    int t = tid + j * 256, r = t >> 5, rem = t & 31, p = rem >> 4, s = rem & 15;
                    const unsigned short* g = (p ? cl : ch) + (size_t)(nbase + r) * C_ + kn + s * 8;
                    async_lds_b128(g, &sB[(kp + 1) & 1][r * LDS_ROW + p * 128 + s * 8]);
                }
                WAIT_ASYNC(8);   // current panel's 8 transfers done; next 8 stay in flight
            } else {
                WAIT_ASYNC(0);
            }
            __syncthreads();

            const unsigned short* pan = &sB[kp & 1][0];
#pragma unroll 1
            for (int ks = 0; ks < KPANEL / 32; ++ks) {   // 4 K-steps of 32
                const int k  = kp * KPANEL + ks * 32;    // global K (A)
                const int kl = ks * 32 + half * 8;       // local K (B, this lane's half)
                v16bf a0h = load_a_frag(zr0h + k);
                v16bf a0l = load_a_frag(zr0l + k);
                v16bf a1h = load_a_frag(zr1h + k);
                v16bf a1l = load_a_frag(zr1l + k);
#pragma unroll
                for (int nt = 0; nt < 4; ++nt) {
                    const unsigned short* bp = pan + (nt * 16 + l16) * LDS_ROW + kl;
                    v16bf bh = load_b_frag(bp);        // hi plane
                    v16bf bl = load_b_frag(bp + 128);  // lo plane
                    acc[0][nt] = wmma_bf16(a0h, bh, acc[0][nt]);
                    acc[0][nt] = wmma_bf16(a0h, bl, acc[0][nt]);
                    acc[0][nt] = wmma_bf16(a0l, bh, acc[0][nt]);
                    acc[1][nt] = wmma_bf16(a1h, bh, acc[1][nt]);
                    acc[1][nt] = wmma_bf16(a1h, bl, acc[1][nt]);
                    acc[1][nt] = wmma_bf16(a1l, bh, acc[1][nt]);
                }
            }
            __syncthreads();   // everyone done reading this buffer before it is restaged
        }

        // dist = ||c||^2 - 2 z.c  (||z||^2 is per-row constant -> irrelevant to argmin)
#pragma unroll
        for (int nt = 0; nt < 4; ++nt) {
            const int n = nbase + nt * 16 + l16;
            const float c2n = c2[n];
#pragma unroll
            for (int r = 0; r < 8; ++r) {
                float d0 = fmaf(-2.0f, acc[0][nt][r], c2n);
                float d1 = fmaf(-2.0f, acc[1][nt][r], c2n);
                if (d0 < minv0[r] || (d0 == minv0[r] && n < mini0[r])) { minv0[r] = d0; mini0[r] = n; }
                if (d1 < minv1[r] || (d1 == minv1[r] && n < mini1[r])) { minv1[r] = d1; mini1[r] = n; }
            }
        }
    }

    // C/D layout: VGPR r, lanes 0-15 -> row M=r, lanes 16-31 -> row M=r+8.
#pragma unroll
    for (int mt = 0; mt < 2; ++mt) {
        const int mtile = mt ? mtile1 : mtile0;
#pragma unroll
        for (int r = 0; r < 8; ++r) {
            float v = mt ? minv1[r] : minv0[r];
            int   i = mt ? mini1[r] : mini0[r];
#pragma unroll
            for (int m = 8; m >= 1; m >>= 1) {
                float ov = __shfl_xor(v, m, 32);
                int   oi = __shfl_xor(i, m, 32);
                if (ov < v || (ov == v && oi < i)) { v = ov; i = oi; }
            }
            if (l16 == 0) {
                const int row = mtile * 16 + r + 8 * half;
                atomicMin(&keys[row], pack_key(v, i));
            }
        }
    }
}

// ---------- kernel 5: gather + mask + outputs ----------
__global__ void vq_gather(const unsigned long long* __restrict__ keys,
                          const float* __restrict__ cb,
                          const unsigned char* __restrict__ mask,
                          float* __restrict__ qout,
                          float* __restrict__ idx_out,
                          float* __restrict__ loss) {
    const int row = blockIdx.x;
    const unsigned long long k = keys[row];
    const int idx = (int)(unsigned)(k & 0xFFFFFFFFu);
    if (threadIdx.x == 0) idx_out[row] = (float)idx;
    if (row == 0 && threadIdx.x == 0) *loss = 0.0f;
    const bool keep = mask[row] != 0;
    const float4* src = reinterpret_cast<const float4*>(cb + (size_t)idx * C_);
    float4* dst = reinterpret_cast<float4*>(qout + (size_t)row * C_);
    float4 v = keep ? src[threadIdx.x] : make_float4(0.f, 0.f, 0.f, 0.f);
    dst[threadIdx.x] = v;
}

// ---------- launch ----------
extern "C" void kernel_launch(void* const* d_in, const int* in_sizes, int n_in,
                              void* d_out, int out_size, void* d_ws, size_t ws_size,
                              hipStream_t stream) {
    const float*         z    = (const float*)d_in[0];          // [8,1024,1024] f32
    const unsigned char* mask = (const unsigned char*)d_in[1];  // [8,1024] bool
    const float*         cb   = (const float*)d_in[2];          // [5000,1024] f32

    char* ws = (char*)d_ws;   // ~54 MB used
    size_t off = 0;
    auto carve = [&](size_t bytes) -> void* {
        void* p = ws + off;
        off += (bytes + 255) & ~(size_t)255;
        return p;
    };
    unsigned short*     zh   = (unsigned short*)carve((size_t)ROWS * C_ * 2);
    unsigned short*     zl   = (unsigned short*)carve((size_t)ROWS * C_ * 2);
    unsigned short*     chh  = (unsigned short*)carve((size_t)NPAD * C_ * 2);
    unsigned short*     cll  = (unsigned short*)carve((size_t)NPAD * C_ * 2);
    float*              c2   = (float*)carve((size_t)NPAD * 4);
    unsigned long long* keys = (unsigned long long*)carve((size_t)ROWS * 8);

    float* qout    = (float*)d_out;
    float* idx_out = qout + (size_t)ROWS * C_;
    float* loss    = idx_out + ROWS;

    vq_init_keys<<<(ROWS + 255) / 256, 256, 0, stream>>>(keys);
    vq_convert_z<<<(ROWS * C_ / 4) / 256, 256, 0, stream>>>(z, zh, zl);
    vq_convert_cb<<<NPAD, 256, 0, stream>>>(cb, chh, cll, c2);
    vq_gemm_argmin<<<(ROWS / 32 / 8) * NSPLIT, 256, 0, stream>>>(zh, zl, chh, cll, c2, keys);
    vq_gather<<<ROWS, 256, 0, stream>>>(keys, cb, mask, qout, idx_out, loss);
}
